// MoDTransformerWrapper_40329742909556
// MI455X (gfx1250) — matrix operand order repaired and checked by reading, not compiled, not run on hardware
//
#include <hip/hip_runtime.h>
#include <math.h>

// Problem sizes (match reference)
#define Bb 4
#define Tt 8192
#define Dd 1024
#define Hh 1024
#define Ll 4
#define Kk 4096   // ceil(0.5 * T)

typedef __attribute__((ext_vector_type(16))) __bf16 v16bf;
typedef __attribute__((ext_vector_type(8)))  float  v8f;

union FragAB { unsigned u[8]; v16bf v; };

__device__ __forceinline__ unsigned short f2bf(float f) {
  unsigned u = __float_as_uint(f);
  u += 0x7FFFu + ((u >> 16) & 1u);          // round-to-nearest-even
  return (unsigned short)(u >> 16);
}

__device__ __forceinline__ float gelu_tanh(float x) {
  const float c0 = 0.7978845608028654f;      // sqrt(2/pi)
  float t = tanhf(c0 * (x + 0.044715f * x * x * x));
  return 0.5f * x * (1.0f + t);
}

// Monotonic map: larger float -> larger unsigned
__device__ __forceinline__ unsigned scoreKey(float s) {
  unsigned u = __float_as_uint(s);
  return (u & 0x80000000u) ? ~u : (u | 0x80000000u);
}

// ---------------------------------------------------------------- scores ----
// One wave per token: scores[b,t] = dot(x[b,t,:], gate[:])
__global__ __launch_bounds__(256) void scores_kernel(
    const float* __restrict__ x, const float* __restrict__ gate,
    float* __restrict__ scores) {
  __shared__ float gl[Dd];
  const int tid = threadIdx.x;
  for (int i = tid; i < Dd; i += 256) gl[i] = gate[i];
  __syncthreads();
  const int wid = tid >> 5, lane = tid & 31;
  const int token = blockIdx.x * 8 + wid;            // 0 .. B*T-1
  const float* xr = x + (size_t)token * Dd;
  float acc = 0.f;
  for (int j = lane; j < Dd; j += 32) acc = fmaf(xr[j], gl[j], acc);
  #pragma unroll
  for (int off = 16; off > 0; off >>= 1) acc += __shfl_xor(acc, off, 32);
  if (lane == 0) scores[token] = acc;
}

// ------------------------------------------------------------------- aux ----
// Deterministic per-(layer,batch) partial: (sigmoid(scores).mean(T) - 0.5)^2
__global__ __launch_bounds__(256) void aux_kernel(
    const float* __restrict__ scores, float* __restrict__ parts) {
  __shared__ float red[256];
  const int b = blockIdx.x, tid = threadIdx.x;
  const float* s = scores + (size_t)b * Tt;
  float sum = 0.f;
  for (int i = tid; i < Tt; i += 256) sum += 1.f / (1.f + expf(-s[i]));
  red[tid] = sum; __syncthreads();
  for (int off = 128; off > 0; off >>= 1) {
    if (tid < off) red[tid] += red[tid + off];
    __syncthreads();
  }
  if (tid == 0) {
    float m = red[0] * (1.0f / Tt) - 0.5f;
    parts[b] = m * m;
  }
}

__global__ void finalize_kernel(const float* __restrict__ parts,
                                float* __restrict__ dst) {
  float a = 0.f;
  for (int i = 0; i < Ll * Bb; ++i) a += parts[i];
  *dst = 0.01f * (a * (1.0f / Bb));
}

// ------------------------------------------------------------- top-K set ----
// One block per batch row. 4-pass radix select over monotonic keys in LDS,
// then emit all strictly-greater indices plus index-ordered ties.
__global__ __launch_bounds__(1024) void topk_kernel(
    const float* __restrict__ scores, int* __restrict__ idxOut) {
  __shared__ unsigned su[Tt];
  __shared__ unsigned hist[256];
  __shared__ unsigned sPrefix, sMask, sRemain, sCnt;
  const int b = blockIdx.x, tid = threadIdx.x;
  const float* s = scores + (size_t)b * Tt;
  for (int i = tid; i < Tt; i += 1024) su[i] = scoreKey(s[i]);
  if (tid == 0) { sPrefix = 0u; sMask = 0u; sRemain = Kk; sCnt = 0u; }
  __syncthreads();
  for (int pass = 0; pass < 4; ++pass) {
    const int shift = 24 - 8 * pass;
    if (tid < 256) hist[tid] = 0u;
    __syncthreads();
    const unsigned prefix = sPrefix, mask = sMask;
    for (int i = tid; i < Tt; i += 1024) {
      unsigned u = su[i];
      if ((u & mask) == prefix) atomicAdd(&hist[(u >> shift) & 255u], 1u);
    }
    __syncthreads();
    if (tid == 0) {
      unsigned rem = sRemain, cum = 0u; int c = 255;
      for (; c > 0; --c) {
        unsigned h = hist[c];
        if (cum + h >= rem) break;
        cum += h;
      }
      sRemain = rem - cum;
      sPrefix = prefix | ((unsigned)c << shift);
      sMask   = mask | (0xFFu << shift);
    }
    __syncthreads();
  }
  const unsigned uT  = sPrefix;   // K-th largest key
  const unsigned nEq = sRemain;   // how many ties to keep
  for (int i = tid; i < Tt; i += 1024) {
    if (su[i] > uT) {
      unsigned p = atomicAdd(&sCnt, 1u);
      idxOut[(size_t)b * Kk + p] = i;
    }
  }
  __syncthreads();
  const unsigned g = sCnt;        // = Kk - nEq
  for (int i = tid; i < Tt; i += 1024) {
    if (su[i] == uT) {
      unsigned rank = 0;
      for (int j = 0; j < i; ++j) rank += (su[j] == uT) ? 1u : 0u;
      if (rank < nEq) idxOut[(size_t)b * Kk + g + rank] = i;
    }
  }
}

// --------------------------------------------------------- gather / cvt ----
__global__ void gather_kernel(const float* __restrict__ x,
                              const int* __restrict__ idx,
                              unsigned short* __restrict__ selA) {
  const size_t total = (size_t)Bb * Kk * Dd;
  for (size_t i = (size_t)blockIdx.x * blockDim.x + threadIdx.x; i < total;
       i += (size_t)gridDim.x * blockDim.x) {
    size_t row = i >> 10;            // /Dd
    int col = (int)(i & (Dd - 1));
    int b = (int)(row >> 12);        // /Kk
    int k = (int)(row & (Kk - 1));
    int tok = idx[b * Kk + k];
    selA[i] = f2bf(x[((size_t)b * Tt + tok) * Dd + col]);
  }
}

__global__ void cvt_bf16_kernel(const float* __restrict__ src,
                                unsigned short* __restrict__ dst, int n) {
  for (int i = blockIdx.x * blockDim.x + threadIdx.x; i < n;
       i += gridDim.x * blockDim.x)
    dst[i] = f2bf(src[i]);
}

// ------------------------------------------------------------- WMMA GEMM ----
#define BM 128
#define BN 128
#define BK 32
#define ASTR 42    // padded ushort stride for A tile (bank-conflict free)
#define BROW 264   // padded ushort stride per K-pair row of packed B tile

// A fragment: 16x32 bf16, ISA layout (lanes 0-15: M, K 0-7/16-23; lanes 16-31: K+8)
__device__ __forceinline__ FragAB loadA_frag(const unsigned short* As,
                                             int rowBase, int lane) {
  const int lo = lane & 15, hi = lane >> 4;
  FragAB r;
  const int base = (rowBase + lo) * ASTR;
  #pragma unroll
  for (int v = 0; v < 8; ++v) {
    int kloc = ((v & 3) << 1) + ((v & 4) << 2) + (hi << 3);
    r.u[v] = *(const unsigned*)&As[base + kloc];
  }
  return r;
}

// B fragment: 32x16 bf16; LDS stores B K-pair packed so each VGPR is one b32 load
__device__ __forceinline__ FragAB loadB_frag(const unsigned short* Bs,
                                             int nBase, int lane) {
  const int lo = lane & 15, hi = lane >> 4;
  FragAB r;
  const int col2 = (nBase + lo) << 1;
  #pragma unroll
  for (int v = 0; v < 8; ++v) {
    int kp = v + (v & 4) + (hi << 2);
    r.u[v] = *(const unsigned*)&Bs[kp * BROW + col2];
  }
  return r;
}

// EPI 0: C = GELU(A*B) -> bf16 mid.   EPI 1: C = A*B -> f32, scattered into x.
template <int EPI>
__global__ __launch_bounds__(256) void gemm_bf16_kernel(
    const unsigned short* __restrict__ A, const unsigned short* __restrict__ Bw,
    unsigned short* __restrict__ midOut, const int* __restrict__ idx,
    float* __restrict__ xOut) {
  __shared__ unsigned short As[BM * ASTR];
  __shared__ unsigned short Bs[16 * BROW];
  const int tid  = threadIdx.x;
  const int lane = tid & 31;
  const int wid  = tid >> 5;
  const int wm = (wid & 3) * 32;   // wave M offset in block tile
  const int wn = (wid >> 2) * 64;  // wave N offset in block tile
  const int Mbase = blockIdx.x * BM;
  const int Nbase = blockIdx.y * BN;

  v8f z; 
  #pragma unroll
  for (int e = 0; e < 8; ++e) z[e] = 0.f;
  v8f acc[2][4];
  #pragma unroll
  for (int i = 0; i < 2; ++i)
    #pragma unroll
    for (int j = 0; j < 4; ++j) acc[i][j] = z;

  for (int kb = 0; kb < Dd; kb += BK) {
    // Stage A tile 128x32 (row-major, uint4 global loads)
    #pragma unroll
    for (int s = 0; s < 2; ++s) {
      int slot = tid + s * 256;
      int row  = slot >> 2;
      int colb = (slot & 3) << 3;
      const uint4 av = *(const uint4*)(A + ((size_t)(Mbase + row) * Dd + kb + colb));
      unsigned* dst = (unsigned*)&As[row * ASTR + colb];
      dst[0] = av.x; dst[1] = av.y; dst[2] = av.z; dst[3] = av.w;
    }
    // Stage B tile 32x128, packed so consecutive K pairs are adjacent ushorts
    #pragma unroll
    for (int s = 0; s < 2; ++s) {
      int slot = tid + s * 256;
      int kr = slot >> 4;
      int nc = (slot & 15) << 3;
      const uint4 bv = *(const uint4*)(Bw + ((size_t)(kb + kr) * Dd + Nbase + nc));
      unsigned short* dst = &Bs[(kr >> 1) * BROW + (nc << 1) + (kr & 1)];
      dst[0]  = (unsigned short)bv.x; dst[2]  = (unsigned short)(bv.x >> 16);
      dst[4]  = (unsigned short)bv.y; dst[6]  = (unsigned short)(bv.y >> 16);
      dst[8]  = (unsigned short)bv.z; dst[10] = (unsigned short)(bv.z >> 16);
      dst[12] = (unsigned short)bv.w; dst[14] = (unsigned short)(bv.w >> 16);
    }
    __syncthreads();

    FragAB af[2], bf[4];
    #pragma unroll
    for (int mt = 0; mt < 2; ++mt) af[mt] = loadA_frag(As, wm + mt * 16, lane);
    #pragma unroll
    for (int nt = 0; nt < 4; ++nt) bf[nt] = loadB_frag(Bs, wn + nt * 16, lane);
    #pragma unroll
    for (int mt = 0; mt < 2; ++mt)
      #pragma unroll
      for (int nt = 0; nt < 4; ++nt)
        acc[mt][nt] = __builtin_amdgcn_wmma_f32_16x16x32_bf16(
            false, af[mt].v, false, bf[nt].v, (short)0, acc[mt][nt], false, false);
    __syncthreads();
  }

  // Epilogue. f32 C layout: lanes 0-15 N=lo M=r; lanes 16-31 N=lo M=r+8.
  const int lo = lane & 15, hi = lane >> 4;
  #pragma unroll
  for (int mt = 0; mt < 2; ++mt)
    #pragma unroll
    for (int nt = 0; nt < 4; ++nt)
      #pragma unroll
      for (int r = 0; r < 8; ++r) {
        int m = Mbase + wm + mt * 16 + r + hi * 8;
        int n = Nbase + wn + nt * 16 + lo;
        float val = acc[mt][nt][r];
        if (EPI == 0) {
          midOut[(size_t)m * Hh + n] = f2bf(gelu_tanh(val));
        } else {
          int b = m >> 12;             // / Kk
          int k = m & (Kk - 1);
          int tok = idx[b * Kk + k];
          xOut[((size_t)b * Tt + tok) * Dd + n] = val;
        }
      }
}

// ---------------------------------------------------------------- launch ----
extern "C" void kernel_launch(void* const* d_in, const int* in_sizes, int n_in,
                              void* d_out, int out_size, void* d_ws, size_t ws_size,
                              hipStream_t stream) {
  const float* x    = (const float*)d_in[0];
  const float* gate = (const float*)d_in[1];
  const float* W1   = (const float*)d_in[2];
  const float* W2   = (const float*)d_in[3];
  float* out = (float*)d_out;   // [B*T*D] working/out x, then [1] aux

  char* ws = (char*)d_ws;
  size_t off = 0;
  auto alloc = [&](size_t bytes) -> void* {
    void* p = ws + off;
    off = (off + bytes + 255) & ~(size_t)255;
    return p;
  };
  float* scores          = (float*)alloc((size_t)Bb * Tt * 4);
  int*   idx             = (int*)alloc((size_t)Bb * Kk * 4);
  float* auxParts        = (float*)alloc((size_t)Ll * Bb * 4);
  unsigned short* W1bf   = (unsigned short*)alloc((size_t)Ll * Dd * Hh * 2);
  unsigned short* W2bf   = (unsigned short*)alloc((size_t)Ll * Hh * Dd * 2);
  unsigned short* selA   = (unsigned short*)alloc((size_t)Bb * Kk * Dd * 2);
  unsigned short* mid    = (unsigned short*)alloc((size_t)Bb * Kk * Hh * 2);

  // Working x = input x (updated in place per layer).
  hipMemcpyAsync(out, x, (size_t)Bb * Tt * Dd * sizeof(float),
                 hipMemcpyDeviceToDevice, stream);

  // One-time bf16 weight conversion (all layers).
  cvt_bf16_kernel<<<4096, 256, 0, stream>>>(W1, W1bf, Ll * Dd * Hh);
  cvt_bf16_kernel<<<4096, 256, 0, stream>>>(W2, W2bf, Ll * Hh * Dd);

  dim3 ggrid(Bb * Kk / BM, Dd / BN);   // 128 x 8

  for (int l = 0; l < Ll; ++l) {
    scores_kernel<<<Bb * Tt / 8, 256, 0, stream>>>(out, gate + (size_t)l * Dd, scores);
    aux_kernel<<<Bb, 256, 0, stream>>>(scores, auxParts + l * Bb);
    topk_kernel<<<Bb, 1024, 0, stream>>>(scores, idx);
    gather_kernel<<<2048, 256, 0, stream>>>(out, idx, selA);
    gemm_bf16_kernel<0><<<ggrid, 256, 0, stream>>>(
        selA, W1bf + (size_t)l * Dd * Hh, mid, idx, out);
    gemm_bf16_kernel<1><<<ggrid, 256, 0, stream>>>(
        mid, W2bf + (size_t)l * Hh * Dd, (unsigned short*)nullptr, idx, out);
  }

  finalize_kernel<<<1, 1, 0, stream>>>(auxParts, out + (size_t)Bb * Tt * Dd);
}
